// Attention_22196390986394
// MI455X (gfx1250) — matrix operand order repaired and checked by reading, not compiled, not run on hardware
//
#include <hip/hip_runtime.h>

// ---------------------------------------------------------------------------
// Causal MHA forward for fixed sizes B=2, T=2048, D=2048, H=16, DH=128.
// ~206 GFLOP vs ~134 MB unique traffic => compute-bound: run every matmul on
// the bf16 WMMA pipe (16x16x32, fp32 accumulate). GEMMs are fed through the
// Tensor Data Mover (tensor_load_to_lds, double-buffered, TENSORcnt-tracked)
// so issue slots go to v_wmma + ds_load instead of per-lane global addressing.
// ---------------------------------------------------------------------------
#define DEVINL __device__ __forceinline__

typedef __attribute__((ext_vector_type(16))) __bf16          bf16x16;
typedef __attribute__((ext_vector_type(8)))  float           f32x8;
typedef __attribute__((ext_vector_type(16))) unsigned short  u16x16;
typedef __attribute__((ext_vector_type(4)))  unsigned int    u32x4;
typedef __attribute__((ext_vector_type(8)))  unsigned int    u32x8;

static constexpr int kB  = 2;
static constexpr int kT  = 2048;
static constexpr int kD  = 2048;
static constexpr int kH  = 16;
static constexpr int kDH = 128;
static constexpr int kM  = kB * kT;          // 4096 rows for the projections

DEVINL f32x8 vzero8() {
    f32x8 z;
#pragma unroll
    for (int i = 0; i < 8; ++i) z[i] = 0.0f;
    return z;
}

DEVINL unsigned short f2bf(float f) {
    unsigned int u = __builtin_bit_cast(unsigned int, f);
    u += 0x7fffu + ((u >> 16) & 1u);          // round-to-nearest-even
    return (unsigned short)(u >> 16);
}

DEVINL f32x8 wmma_bf16(bf16x16 a, bf16x16 b, f32x8 c) {
    return __builtin_amdgcn_wmma_f32_16x16x32_bf16(false, a, false, b, (short)0, c,
                                                   false, false);
}

// A-fragment (16x32 bf16, MxK): lane l holds row M=l&15.
// lanes 0-15 -> K = {0..7, 16..23}; lanes 16-31 -> K = {8..15, 24..31}.
DEVINL bf16x16 load_frag_a(const unsigned short* base, int ld) {
    const int l = threadIdx.x & 31;
    const unsigned short* p = base + (size_t)(l & 15) * ld + ((l < 16) ? 0 : 8);
    u16x16 u;
#pragma unroll
    for (int i = 0; i < 8; ++i) u[i] = p[i];
#pragma unroll
    for (int i = 0; i < 8; ++i) u[8 + i] = p[16 + i];
    return __builtin_bit_cast(bf16x16, u);
}

// B-fragment (32x16 bf16, KxN) read from an N-major buffer (row = N, contig K):
// lane l holds column N=l&15; lanes 0-15 -> K = {0..7, 16..23}; lanes 16-31 -> K = {8..15, 24..31}.
DEVINL bf16x16 load_frag_b(const unsigned short* base, int ld) {
    const int l = threadIdx.x & 31;
    const unsigned short* p = base + (size_t)(l & 15) * ld + ((l < 16) ? 0 : 8);
    u16x16 u;
#pragma unroll
    for (int i = 0; i < 8; ++i) u[i] = p[i];
#pragma unroll
    for (int i = 0; i < 8; ++i) u[8 + i] = p[16 + i];
    return __builtin_bit_cast(bf16x16, u);
}

// ---------------------------------------------------------------------------
// Tensor Data Mover: 2D tile load Global -> LDS (CDNA5 ISA).
// D# group0: count=1 | lds byte addr | 57b global addr | type=2 ("image").
// D# group1: data_size=2B, pad_enable, pad_interval=16 DWORDs (one 32-elem row),
//            pad_amount=4 DWORDs  => LDS row pitch 80 B (ld = 40 elements),
//            which de-conflicts the ds_load_b128 fragment reads.
// Inline asm form is portable across both toolchains (bypasses builtin arity).
// ---------------------------------------------------------------------------
DEVINL void tdm_load_2d(unsigned lds_byte_addr, const unsigned short* gtile,
                        unsigned tensor_d0, unsigned tensor_d1,
                        unsigned tile_d0, unsigned tile_d1, unsigned stride0) {
    unsigned long long ga = (unsigned long long)gtile;
    u32x4 g0;
    g0[0] = 1u;                                            // count=1, user D#
    g0[1] = lds_byte_addr;
    g0[2] = (unsigned)ga;
    g0[3] = (unsigned)((ga >> 32) & 0x1FFFFFFu) | (2u << 30);
    u32x8 g1;
    g1[0] = (1u << 16)      // data_size: 2 bytes
          | (1u << 20)      // pad_enable
          | (3u << 22)      // pad_interval: 16 DWORDs
          | (3u << 25);     // pad_amount:   4 DWORDs
    g1[1] = (tensor_d0 & 0xFFFFu) << 16;                   // tensor_dim0 lo
    g1[2] = (tensor_d0 >> 16) | ((tensor_d1 & 0xFFFFu) << 16);
    g1[3] = (tensor_d1 >> 16) | (tile_d0 << 16);
    g1[4] = tile_d1;                                       // tile_dim1 (tile_dim2=0)
    g1[5] = stride0;                                       // tensor_dim0_stride lo
    g1[6] = 0;
    g1[7] = 0;
    asm volatile("tensor_load_to_lds %0, %1" : : "s"(g0), "s"(g1) : "memory");
}

// ---------------------------------------------------------------------------
// Stage 1a: fp32 -> bf16 elementwise
// ---------------------------------------------------------------------------
__global__ __launch_bounds__(256) void cvt_bf16_kernel(const float* __restrict__ in,
                                                       unsigned short* __restrict__ out,
                                                       int n) {
    int i = blockIdx.x * 256 + threadIdx.x;
    if (i < n) out[i] = f2bf(in[i]);
}

// ---------------------------------------------------------------------------
// Stage 1b: fp32 (KxN) -> bf16 transposed (NxK), 2048x2048, tiled via LDS
// ---------------------------------------------------------------------------
__global__ __launch_bounds__(256) void cvt_transpose_kernel(const float* __restrict__ in,
                                                            unsigned short* __restrict__ out) {
    constexpr int N = 2048;
    __shared__ unsigned short tile[32][33];
    const int tx = threadIdx.x & 31;
    const int ty = threadIdx.x >> 5;
    const int bx = blockIdx.x * 32;             // n
    const int by = blockIdx.y * 32;             // k
#pragma unroll
    for (int i = 0; i < 32; i += 8)
        tile[ty + i][tx] = f2bf(in[(size_t)(by + ty + i) * N + (bx + tx)]);
    __syncthreads();
#pragma unroll
    for (int i = 0; i < 32; i += 8)
        out[(size_t)(bx + ty + i) * N + (by + tx)] = tile[tx][ty + i];
}

// ---------------------------------------------------------------------------
// Stage 2/4: BF16 WMMA GEMM, TDM-fed.  C(4096,2048) = A(4096,2048) x Bt(2048,2048)^T
//   A  : bf16 row-major (M x K);  Bt : bf16 row-major (N x K)
// Block = 256 threads = 8 waves (2x4); wave tile 64x64; block tile 128x256.
// Per 32-deep k-step: TDM stages A(128x32) + Bt(256x32) into padded LDS
// (double-buffered, wave 0 drives TENSORcnt), each wave does 16 WMMAs fed by
// 16 ds_load_b128 fragment reads -> 1:1 load:wmma.
// ---------------------------------------------------------------------------
template <bool OUT_F32, bool TRANS>
__global__ __launch_bounds__(256) void gemm_bf16_kernel(const unsigned short* __restrict__ A,
                                                        const unsigned short* __restrict__ Bt,
                                                        void* __restrict__ Out) {
    constexpr int LDA    = 40;    // padded LDS pitch in elements (80 B)
    constexpr int A_ROWS = 128;
    constexpr int B_ROWS = 256;
    __shared__ unsigned short smem[(2 * A_ROWS + 2 * B_ROWS) * LDA];   // 60 KB

    const int wave = threadIdx.x >> 5;
    const int wrow = wave >> 2;                  // 0..1
    const int wcol = wave & 3;                   // 0..3
    const int mblk = blockIdx.y * 128;
    const int nblk = blockIdx.x * 256;
    const int m0 = mblk + wrow * 64;
    const int n0 = nblk + wcol * 64;

    const unsigned ldsBase = (unsigned)(unsigned long long)(const void*)smem;
    const unsigned aOff[2] = {0u, (unsigned)(A_ROWS * LDA * 2)};
    const unsigned bOff[2] = {(unsigned)(2 * A_ROWS * LDA * 2),
                              (unsigned)(2 * A_ROWS * LDA * 2 + B_ROWS * LDA * 2)};

    f32x8 acc[4][4];
#pragma unroll
    for (int i = 0; i < 4; ++i)
#pragma unroll
        for (int j = 0; j < 4; ++j) acc[i][j] = vzero8();

    // Prologue: stage k-slice 0 into buffer 0.
    if (threadIdx.x < 32) {
        tdm_load_2d(ldsBase + aOff[0], A + (size_t)mblk * kD, kD, kM, 32, A_ROWS, kD);
        tdm_load_2d(ldsBase + bOff[0], Bt + (size_t)nblk * kD, kD, kD, 32, B_ROWS, kD);
    }

    int buf = 0;
    for (int k0 = 0; k0 < kD; k0 += 32, buf ^= 1) {
        const bool more = (k0 + 32) < kD;
        if (threadIdx.x < 32) {
            if (more) {
                tdm_load_2d(ldsBase + aOff[buf ^ 1], A + (size_t)mblk * kD + (k0 + 32),
                            kD, kM, 32, A_ROWS, kD);
                tdm_load_2d(ldsBase + bOff[buf ^ 1], Bt + (size_t)nblk * kD + (k0 + 32),
                            kD, kD, 32, B_ROWS, kD);
                __builtin_amdgcn_s_wait_tensorcnt(2);   // current buffer ready
            } else {
                __builtin_amdgcn_s_wait_tensorcnt(0);
            }
        }
        __syncthreads();

        const unsigned short* As = smem + buf * A_ROWS * LDA;
        const unsigned short* Bs = smem + 2 * A_ROWS * LDA + buf * B_ROWS * LDA;

        bf16x16 af[4], bfv[4];
#pragma unroll
        for (int mi = 0; mi < 4; ++mi)
            af[mi] = load_frag_a(As + (wrow * 64 + mi * 16) * LDA, LDA);
#pragma unroll
        for (int ni = 0; ni < 4; ++ni)
            bfv[ni] = load_frag_b(Bs + (wcol * 64 + ni * 16) * LDA, LDA);
#pragma unroll
        for (int mi = 0; mi < 4; ++mi)
#pragma unroll
            for (int ni = 0; ni < 4; ++ni)
                acc[mi][ni] = wmma_bf16(af[mi], bfv[ni], acc[mi][ni]);

        __syncthreads();    // all reads done before next TDM overwrites `buf`
    }

    // C/D layout: lane l holds N=l&15; VGPR r holds M = r + (l<16 ? 0 : 8).
    const int lane = threadIdx.x & 31;
    const int cn = lane & 15;
    const int cm = (lane < 16) ? 0 : 8;
#pragma unroll
    for (int mi = 0; mi < 4; ++mi)
#pragma unroll
        for (int ni = 0; ni < 4; ++ni) {
            f32x8 a = acc[mi][ni];
            const int mbase = m0 + mi * 16 + cm;
            const int nglob = n0 + ni * 16 + cn;
#pragma unroll
            for (int r = 0; r < 8; ++r) {
                const int m = mbase + r;
                if (OUT_F32) {
                    ((float*)Out)[(size_t)m * kD + nglob] = a[r];
                } else if (!TRANS) {
                    ((unsigned short*)Out)[(size_t)m * kD + nglob] = f2bf(a[r]);
                } else {
                    const int bb = m >> 11;         // m / T
                    const int tt = m & (kT - 1);    // m % T
                    ((unsigned short*)Out)[((size_t)bb * kD + nglob) * kT + tt] =
                        f2bf(a[r]);
                }
            }
        }
}

// ---------------------------------------------------------------------------
// Stage 3: flash attention (causal), one wave per 16-row q-tile.
//   Q : bf16 (B, T, D);  K : bf16 (B, T, D);  Vt: bf16 (B, D, T);  ctx: bf16 (B,T,D)
// ---------------------------------------------------------------------------
__global__ __launch_bounds__(128) void flash_attn_kernel(const unsigned short* __restrict__ Q,
                                                         const unsigned short* __restrict__ K,
                                                         const unsigned short* __restrict__ Vt,
                                                         unsigned short* __restrict__ ctx) {
    constexpr float kScale = 0.08838834764831845f;   // 1/sqrt(128)
    constexpr float kNegBig = -3.0e38f;

    __shared__ unsigned short plds[4][16 * 32];       // 1 KB per wave

    const int wave = threadIdx.x >> 5;
    const int lane = threadIdx.x & 31;
    const int bh = blockIdx.y;
    const int b = bh >> 4;
    const int h = bh & 15;
    const int qbase = (blockIdx.x * 4 + wave) * 16;

    const unsigned short* Qbase = Q + ((size_t)(b * kT + qbase)) * kD + h * kDH;
    bf16x16 qf[4];
#pragma unroll
    for (int kk = 0; kk < 4; ++kk) qf[kk] = load_frag_a(Qbase + kk * 32, kD);

    f32x8 o[8];
#pragma unroll
    for (int j = 0; j < 8; ++j) o[j] = vzero8();
    float m_i[8], l_i[8];
#pragma unroll
    for (int r = 0; r < 8; ++r) { m_i[r] = kNegBig; l_i[r] = 0.0f; }

    const int cn = lane & 15;
    const int cm = (lane < 16) ? 0 : 8;
    const int kend = qbase + 16;

    const unsigned short* Vhead = Vt + ((size_t)(b * kD + h * kDH)) * kT;

    for (int kb = 0; kb < kend; kb += 32) {
        const unsigned short* Kb0 = K + ((size_t)(b * kT + kb)) * kD + h * kDH;
        if (kb + 32 < kend) {   // speculative prefetch of the next key block
            __builtin_prefetch(Kb0 + (size_t)32 * kD, 0, 0);
            __builtin_prefetch(Vhead + kb + 32, 0, 0);
        }

        // ---- scores S = Q * K^T for 32 keys (two 16-wide n-tiles) ----
        f32x8 s0 = vzero8(), s1 = vzero8();
#pragma unroll
        for (int kk = 0; kk < 4; ++kk) {
            bf16x16 kf0 = load_frag_b(Kb0 + kk * 32, kD);
            bf16x16 kf1 = load_frag_b(Kb0 + (size_t)16 * kD + kk * 32, kD);
            s0 = wmma_bf16(qf[kk], kf0, s0);
            s1 = wmma_bf16(qf[kk], kf1, s1);
        }

        // ---- scale + causal mask ----
#pragma unroll
        for (int r = 0; r < 8; ++r) {
            const int row = qbase + cm + r;
            const float v0 = s0[r] * kScale;
            const float v1 = s1[r] * kScale;
            s0[r] = (kb + cn <= row) ? v0 : kNegBig;
            s1[r] = (kb + 16 + cn <= row) ? v1 : kNegBig;
        }

        // ---- online softmax update (row stats live per 16-lane half) ----
#pragma unroll
        for (int r = 0; r < 8; ++r) {
            float mx = fmaxf(s0[r], s1[r]);
            mx = fmaxf(mx, __shfl_xor(mx, 1, 16));
            mx = fmaxf(mx, __shfl_xor(mx, 2, 16));
            mx = fmaxf(mx, __shfl_xor(mx, 4, 16));
            mx = fmaxf(mx, __shfl_xor(mx, 8, 16));
            const float newm = fmaxf(m_i[r], mx);
            const float alpha = __expf(m_i[r] - newm);
            m_i[r] = newm;
            const float p0 = __expf(s0[r] - newm);
            const float p1 = __expf(s1[r] - newm);
            s0[r] = p0;
            s1[r] = p1;
            float rs = p0 + p1;
            rs += __shfl_xor(rs, 1, 16);
            rs += __shfl_xor(rs, 2, 16);
            rs += __shfl_xor(rs, 4, 16);
            rs += __shfl_xor(rs, 8, 16);
            l_i[r] = l_i[r] * alpha + rs;
#pragma unroll
            for (int j = 0; j < 8; ++j) o[j][r] *= alpha;
        }

        // ---- P (fp32, C-layout) -> bf16 A-fragment via per-wave LDS tile ----
        unsigned short* pl = plds[wave];
#pragma unroll
        for (int r = 0; r < 8; ++r) {
            pl[(cm + r) * 32 + cn]      = f2bf(s0[r]);
            pl[(cm + r) * 32 + 16 + cn] = f2bf(s1[r]);
        }
        __builtin_amdgcn_wave_barrier();
        bf16x16 pf = load_frag_a(pl, 32);

        // ---- O += P * V  (8 n-tiles across d_head=128) ----
        const unsigned short* Vb = Vhead + kb;
#pragma unroll
        for (int j = 0; j < 8; ++j) {
            bf16x16 vf = load_frag_b(Vb + (size_t)(j * 16) * kT, kT);
            o[j] = wmma_bf16(pf, vf, o[j]);
        }
        __builtin_amdgcn_wave_barrier();
    }

    // ---- normalize and write ctx (bf16) ----
    float inv[8];
#pragma unroll
    for (int r = 0; r < 8; ++r) inv[r] = 1.0f / l_i[r];
    unsigned short* cbase = ctx + ((size_t)(b * kT + qbase)) * kD + h * kDH;
#pragma unroll
    for (int j = 0; j < 8; ++j)
#pragma unroll
        for (int r = 0; r < 8; ++r)
            cbase[(size_t)(cm + r) * kD + (j * 16 + cn)] = f2bf(o[j][r] * inv[r]);
}

// ---------------------------------------------------------------------------
// Host-side orchestration
// ---------------------------------------------------------------------------
extern "C" void kernel_launch(void* const* d_in, const int* in_sizes, int n_in,
                              void* d_out, int out_size, void* d_ws, size_t ws_size,
                              hipStream_t stream) {
    (void)in_sizes; (void)n_in; (void)out_size; (void)ws_size;

    const float* x  = (const float*)d_in[0];
    const float* wq = (const float*)d_in[1];
    const float* wk = (const float*)d_in[2];
    const float* wv = (const float*)d_in[3];
    const float* wo = (const float*)d_in[4];

    char* ws = (char*)d_ws;
    size_t off = 0;
    auto carve = [&](size_t bytes) {
        void* p = ws + off;
        off += (bytes + 255) & ~(size_t)255;
        return p;
    };
    const size_t xdBytes = (size_t)kM * kD * sizeof(unsigned short);   // 16 MB
    const size_t wBytes  = (size_t)kD * kD * sizeof(unsigned short);   //  8 MB

    unsigned short* xb  = (unsigned short*)carve(xdBytes);
    unsigned short* wqT = (unsigned short*)carve(wBytes);
    unsigned short* wkT = (unsigned short*)carve(wBytes);
    unsigned short* wvT = (unsigned short*)carve(wBytes);
    unsigned short* woT = (unsigned short*)carve(wBytes);
    unsigned short* Qb  = (unsigned short*)carve(xdBytes);
    unsigned short* Kb  = (unsigned short*)carve(xdBytes);
    unsigned short* Vt  = (unsigned short*)carve(xdBytes);  // (B, D, T)
    unsigned short* Cx  = (unsigned short*)carve(xdBytes);

    {
        const int n = kM * kD;
        cvt_bf16_kernel<<<n / 256, 256, 0, stream>>>(x, xb, n);
        dim3 tg(kD / 32, kD / 32);
        cvt_transpose_kernel<<<tg, 256, 0, stream>>>(wq, wqT);
        cvt_transpose_kernel<<<tg, 256, 0, stream>>>(wk, wkT);
        cvt_transpose_kernel<<<tg, 256, 0, stream>>>(wv, wvT);
        cvt_transpose_kernel<<<tg, 256, 0, stream>>>(wo, woT);
    }

    dim3 ggrid(kD / 256, kM / 128);                              // (8, 32)
    gemm_bf16_kernel<false, false><<<ggrid, 256, 0, stream>>>(xb, wqT, Qb);
    gemm_bf16_kernel<false, false><<<ggrid, 256, 0, stream>>>(xb, wkT, Kb);
    gemm_bf16_kernel<false, true ><<<ggrid, 256, 0, stream>>>(xb, wvT, Vt);

    dim3 agrid(kT / 16 / 4, kB * kH);                            // (32, 32)
    flash_attn_kernel<<<agrid, 128, 0, stream>>>(Qb, Kb, Vt, Cx);

    gemm_bf16_kernel<true, false><<<ggrid, 256, 0, stream>>>(Cx, woT, d_out);
}
